// LoRAGroupedExperts_42949673290
// MI455X (gfx1250) — matrix operand order repaired and checked by reading, not compiled, not run on hardware
//
#include <hip/hip_runtime.h>
#include <hip/hip_bf16.h>

// MI455X / gfx1250: wave32, bf16 WMMA 16x16x32 (fp32 accum), async LDS copies.

typedef __attribute__((ext_vector_type(16))) __bf16 v16bf;
typedef __attribute__((ext_vector_type(8)))  __bf16 v8bf;
typedef __attribute__((ext_vector_type(8)))  float  v8f;

#define E_ 8
#define D_ 2048
#define H_ 4096
#define R_ 16
#define T_ 1024
#define SCALE 2.0f   // ALPHA / R = 32 / 16

#define BM 64
#define BN 128
#define BK 64

static __device__ __forceinline__ __bf16 f2bf(float f) { return (__bf16)f; }

// pack two floats -> two bf16 in one dword (lowers to v_cvt_pk_bf16_f32)
static __device__ __forceinline__ unsigned pk2(float lo, float hi) {
  union { __bf16 b[2]; unsigned u; } t;
  t.b[0] = (__bf16)lo; t.b[1] = (__bf16)hi;
  return t.u;
}

static __device__ __forceinline__ v8f wmma_bf16(v16bf a, v16bf b, v8f c) {
  return __builtin_amdgcn_wmma_f32_16x16x32_bf16(false, a, false, b, (short)0, c,
                                                 false, false);
}

// Swizzled LDS element index. Granule = 16 elements (32B) = one WMMA K-chunk,
// so fragment loads stay contiguous & 16B-aligned while stores spread banks.
static __device__ __forceinline__ int aidx(int r, int k) {   // A tile [BM][BK]
  return r * BK + ((((k >> 4) ^ ((r >> 1) & 3)) << 4) | (k & 15));
}
static __device__ __forceinline__ int bidx(int n, int k) {   // B tile [BN][BK] (N-major)
  return n * BK + ((((k >> 4) ^ ((n >> 2) & 3)) << 4) | (k & 15));
}

// A fragment (16x32): lane-group g holds K {ks+8g..+7} U {ks+16+8g..+7}
static __device__ __forceinline__ v16bf ld_a(const __bf16* ls, int row, int ks, int g) {
  v8bf lo = *(const v8bf*)(ls + aidx(row, ks + 8 * g));
  v8bf hi = *(const v8bf*)(ls + aidx(row, ks + 16 + 8 * g));
  return __builtin_shufflevector(lo, hi, 0,1,2,3,4,5,6,7,8,9,10,11,12,13,14,15);
}
// B fragment (32x16): lane-group g holds K {ks+8g..+7} U {ks+16+8g..+7}
static __device__ __forceinline__ v16bf ld_b(const __bf16* ls, int col, int ks, int g) {
  v8bf lo = *(const v8bf*)(ls + bidx(col, ks + 8 * g));
  v8bf hi = *(const v8bf*)(ls + bidx(col, ks + 16 + 8 * g));
  return __builtin_shufflevector(lo, hi, 0,1,2,3,4,5,6,7,8,9,10,11,12,13,14,15);
}

// ---------------------------------------------------------------------------
// Kernel 0: xa_{gate,up}[e,t,r] = SCALE * (x @ lora_a)   (skinny rank-16 GEMM)
// ---------------------------------------------------------------------------
__global__ __launch_bounds__(256) void lora_xa_kernel(
    const float* __restrict__ x,   // [E*T, D]
    const float* __restrict__ Ag,  // [E, D, R]
    const float* __restrict__ Au,  // [E, D, R]
    float* __restrict__ xag, float* __restrict__ xau)  // [E*T, R]
{
  int tid  = threadIdx.x;
  int lane = tid & 31;
  int gw   = blockIdx.x * 8 + (tid >> 5);   // 0..511
  int e    = gw >> 6;
  int t0   = (gw & 63) * 16;
  int r = lane & 15, g = lane >> 4;

  const float* xrow = x  + ((size_t)e * T_ + t0 + r) * D_;
  const float* ag   = Ag + (size_t)e * D_ * R_;
  const float* au   = Au + (size_t)e * D_ * R_;

  v8f accg = {}; v8f accu = {};
  for (int k0 = 0; k0 < D_; k0 += 32) {
    v16bf af;
    float4 u0 = *(const float4*)(xrow + k0 + g * 8);
    float4 u1 = *(const float4*)(xrow + k0 + g * 8 + 4);
    float4 u2 = *(const float4*)(xrow + k0 + 16 + g * 8);
    float4 u3 = *(const float4*)(xrow + k0 + 16 + g * 8 + 4);
    af[0]=f2bf(u0.x); af[1]=f2bf(u0.y); af[2]=f2bf(u0.z); af[3]=f2bf(u0.w);
    af[4]=f2bf(u1.x); af[5]=f2bf(u1.y); af[6]=f2bf(u1.z); af[7]=f2bf(u1.w);
    af[8]=f2bf(u2.x); af[9]=f2bf(u2.y); af[10]=f2bf(u2.z); af[11]=f2bf(u2.w);
    af[12]=f2bf(u3.x); af[13]=f2bf(u3.y); af[14]=f2bf(u3.z); af[15]=f2bf(u3.w);

    v16bf bg, bu;
#pragma unroll
    for (int q = 0; q < 16; ++q) {
      int k = k0 + (q < 8 ? g * 8 + q : 8 + g * 8 + q);
      bg[q] = f2bf(ag[(size_t)k * R_ + r]);
      bu[q] = f2bf(au[(size_t)k * R_ + r]);
    }
    accg = wmma_bf16(af, bg, accg);
    accu = wmma_bf16(af, bu, accu);
  }
#pragma unroll
  for (int vg = 0; vg < 8; ++vg) {
    int tok = e * T_ + t0 + vg + 8 * g;
    xag[(size_t)tok * R_ + r] = SCALE * accg[vg];
    xau[(size_t)tok * R_ + r] = SCALE * accu[vg];
  }
}

// ---------------------------------------------------------------------------
// Kernel 1: h = silu(xWg + xag@Bg) * (xWu + xau@Bu), stored bf16.
// ---------------------------------------------------------------------------
__global__ __launch_bounds__(256) void gateup_kernel(
    const float* __restrict__ x,
    const float* __restrict__ Wg, const float* __restrict__ Wu,   // [E, D, H]
    const float* __restrict__ Bg, const float* __restrict__ Bu,   // [E, R, H]
    const float* __restrict__ xag, const float* __restrict__ xau, // [E*T, R]
    __bf16* __restrict__ h)                                       // [E*T, H]
{
  __shared__ alignas(32) __bf16 lsA[BM * BK];
  __shared__ alignas(32) __bf16 lsBg[BN * BK];
  __shared__ alignas(32) __bf16 lsBu[BN * BK];

  int tid = threadIdx.x, lane = tid & 31, w = tid >> 5;
  int wm = w >> 2, wn = w & 3;
  int g = lane >> 4, ln = lane & 15;
  int e = blockIdx.z;
  int tokBase = blockIdx.y * BM;
  int n0 = blockIdx.x * BN;

  const float* xe = x  + (size_t)e * T_ * D_;
  const float* wg = Wg + (size_t)e * D_ * H_;
  const float* wu = Wu + (size_t)e * D_ * H_;

  v8f accg[2][2] = {}; v8f accu[2][2] = {};

  for (int k0 = 0; k0 < D_; k0 += BK) {
    __syncthreads();
    // stage A: 64x64 fp32 -> bf16 (packed b64 stores, swizzled)
#pragma unroll
    for (int j = 0; j < 4; ++j) {
      int idx = tid + j * 256;
      int row = idx >> 4;
      int c4  = (idx & 15) * 4;
      float4 v = *(const float4*)(xe + (size_t)(tokBase + row) * D_ + k0 + c4);
      uint2 wv; wv.x = pk2(v.x, v.y); wv.y = pk2(v.z, v.w);
      *(uint2*)(lsA + aidx(row, c4)) = wv;
    }
    // stage B (both weights): wave covers 8 k-pairs x 4 n-quads -> conflict-free
    // packed b32 stores of (k,k+1) pairs into N-major swizzled tiles.
#pragma unroll
    for (int j = 0; j < 4; ++j) {
      int idx   = tid + j * 256;          // bits: [1:0]=n4w [4:2]=kpw [6:5]=kpblk [9:7]=n4blk
      int n4w   = idx & 3;
      int kpw   = (idx >> 2) & 7;
      int kpblk = (idx >> 5) & 3;
      int n4blk = idx >> 7;
      int k = 2 * (kpw + 8 * kpblk);      // even
      int n = 4 * (n4w + 4 * n4blk);
      size_t off = (size_t)(k0 + k) * H_ + n0 + n;
      float4 a0 = *(const float4*)(wg + off);
      float4 a1 = *(const float4*)(wg + off + H_);
      float4 b0 = *(const float4*)(wu + off);
      float4 b1 = *(const float4*)(wu + off + H_);
      if (k0 + BK < D_) {
        __builtin_prefetch(wg + off + (size_t)BK * H_, 0, 0);
        __builtin_prefetch(wu + off + (size_t)BK * H_, 0, 0);
      }
      *(unsigned*)(lsBg + bidx(n + 0, k)) = pk2(a0.x, a1.x);
      *(unsigned*)(lsBg + bidx(n + 1, k)) = pk2(a0.y, a1.y);
      *(unsigned*)(lsBg + bidx(n + 2, k)) = pk2(a0.z, a1.z);
      *(unsigned*)(lsBg + bidx(n + 3, k)) = pk2(a0.w, a1.w);
      *(unsigned*)(lsBu + bidx(n + 0, k)) = pk2(b0.x, b1.x);
      *(unsigned*)(lsBu + bidx(n + 1, k)) = pk2(b0.y, b1.y);
      *(unsigned*)(lsBu + bidx(n + 2, k)) = pk2(b0.z, b1.z);
      *(unsigned*)(lsBu + bidx(n + 3, k)) = pk2(b0.w, b1.w);
    }
    __syncthreads();
#pragma unroll
    for (int ks = 0; ks < BK; ks += 32) {
      v16bf a0 = ld_a(lsA, wm * 32 + ln,      ks, g);
      v16bf a1 = ld_a(lsA, wm * 32 + 16 + ln, ks, g);
#pragma unroll
      for (int ni = 0; ni < 2; ++ni) {
        v16bf bgf = ld_b(lsBg, wn * 32 + ni * 16 + ln, ks, g);
        v16bf buf = ld_b(lsBu, wn * 32 + ni * 16 + ln, ks, g);
        accg[0][ni] = wmma_bf16(a0, bgf, accg[0][ni]);
        accg[1][ni] = wmma_bf16(a1, bgf, accg[1][ni]);
        accu[0][ni] = wmma_bf16(a0, buf, accu[0][ni]);
        accu[1][ni] = wmma_bf16(a1, buf, accu[1][ni]);
      }
    }
  }

  // ---- LoRA rank stage: one zero-padded K=32 step ----
  const float* bgp = Bg + (size_t)e * R_ * H_;
  const float* bup = Bu + (size_t)e * R_ * H_;
  const float* xg  = xag + ((size_t)e * T_ + tokBase) * R_;
  const float* xu  = xau + ((size_t)e * T_ + tokBase) * R_;

  __syncthreads();
#pragma unroll
  for (int j = 0; j < 16; ++j) {           // 128n x 32k
    int idx = tid + j * 256;
    int nn = idx >> 5, kk = idx & 31;
    float a = (kk < R_) ? bgp[(size_t)kk * H_ + n0 + nn] : 0.0f;
    float b = (kk < R_) ? bup[(size_t)kk * H_ + n0 + nn] : 0.0f;
    lsBg[bidx(nn, kk)] = f2bf(a);
    lsBu[bidx(nn, kk)] = f2bf(b);
  }
#pragma unroll
  for (int j = 0; j < 8; ++j) {            // 64rows x 32k : xa_gate
    int idx = tid + j * 256;
    int row = idx >> 5, kk = idx & 31;
    lsA[aidx(row, kk)] = f2bf(kk < R_ ? xg[row * R_ + kk] : 0.0f);
  }
  __syncthreads();
  {
    v16bf a0 = ld_a(lsA, wm * 32 + ln,      0, g);
    v16bf a1 = ld_a(lsA, wm * 32 + 16 + ln, 0, g);
#pragma unroll
    for (int ni = 0; ni < 2; ++ni) {
      v16bf b = ld_b(lsBg, wn * 32 + ni * 16 + ln, 0, g);
      accg[0][ni] = wmma_bf16(a0, b, accg[0][ni]);
      accg[1][ni] = wmma_bf16(a1, b, accg[1][ni]);
    }
  }
  __syncthreads();
#pragma unroll
  for (int j = 0; j < 8; ++j) {            // refill A with xa_up
    int idx = tid + j * 256;
    int row = idx >> 5, kk = idx & 31;
    lsA[aidx(row, kk)] = f2bf(kk < R_ ? xu[row * R_ + kk] : 0.0f);
  }
  __syncthreads();
  {
    v16bf a0 = ld_a(lsA, wm * 32 + ln,      0, g);
    v16bf a1 = ld_a(lsA, wm * 32 + 16 + ln, 0, g);
#pragma unroll
    for (int ni = 0; ni < 2; ++ni) {
      v16bf b = ld_b(lsBu, wn * 32 + ni * 16 + ln, 0, g);
      accu[0][ni] = wmma_bf16(a0, b, accu[0][ni]);
      accu[1][ni] = wmma_bf16(a1, b, accu[1][ni]);
    }
  }

  // ---- epilogue: h = silu(g) * u, bf16 ----
#pragma unroll
  for (int mi = 0; mi < 2; ++mi)
#pragma unroll
    for (int ni = 0; ni < 2; ++ni)
#pragma unroll
      for (int vg = 0; vg < 8; ++vg) {
        int m   = vg + 8 * g;
        int tok = tokBase + wm * 32 + mi * 16 + m;
        int col = n0 + wn * 32 + ni * 16 + ln;
        float gv = accg[mi][ni][vg];
        float uv = accu[mi][ni][vg];
        float sg = gv / (1.0f + __expf(-gv));
        h[((size_t)e * T_ + tok) * H_ + col] = f2bf(sg * uv);
      }
}

// ---------------------------------------------------------------------------
// Kernel 2: ha[e,t,r] = SCALE * (h @ lora_down_a)  (bf16 A, skinny GEMM)
// ---------------------------------------------------------------------------
__global__ __launch_bounds__(256) void lora_ha_kernel(
    const __bf16* __restrict__ h,   // [E*T, H]
    const float* __restrict__ Ad,   // [E, H, R]
    float* __restrict__ ha)         // [E*T, R]
{
  int tid = threadIdx.x, lane = tid & 31;
  int gw = blockIdx.x * 8 + (tid >> 5);
  int e  = gw >> 6;
  int t0 = (gw & 63) * 16;
  int r = lane & 15, g = lane >> 4;

  const __bf16* hrow = h + ((size_t)e * T_ + t0 + r) * H_;
  const float*  ad   = Ad + (size_t)e * H_ * R_;

  v8f acc = {};
  for (int k0 = 0; k0 < H_; k0 += 32) {
    v8bf lo = *(const v8bf*)(hrow + k0 + g * 8);
    v8bf hi = *(const v8bf*)(hrow + k0 + 16 + g * 8);
    v16bf af = __builtin_shufflevector(lo, hi, 0,1,2,3,4,5,6,7,8,9,10,11,12,13,14,15);
    v16bf bf_;
#pragma unroll
    for (int q = 0; q < 16; ++q)
      bf_[q] = f2bf(ad[(size_t)(k0 + (q < 8 ? g * 8 + q : 8 + g * 8 + q)) * R_ + r]);
    acc = wmma_bf16(af, bf_, acc);
  }
#pragma unroll
  for (int vg = 0; vg < 8; ++vg) {
    int tok = e * T_ + t0 + vg + 8 * g;
    ha[(size_t)tok * R_ + r] = SCALE * acc[vg];
  }
}

// ---------------------------------------------------------------------------
// Kernel 3: out = h@Wd + ha@Bd, fp32 output. A tile is a raw bf16 copy ->
// use gfx1250 async global->LDS DMA (ASYNCcnt) instead of VGPR round-trip.
// ---------------------------------------------------------------------------
__global__ __launch_bounds__(256) void down_kernel(
    const __bf16* __restrict__ h,   // [E*T, H]
    const float* __restrict__ Wd,   // [E, H, D]
    const float* __restrict__ Bd,   // [E, R, D]
    const float* __restrict__ ha,   // [E*T, R]
    float* __restrict__ out)        // [E*T, D]
{
  __shared__ alignas(32) __bf16 lsA[BM * BK];
  __shared__ alignas(32) __bf16 lsB[BN * BK];

  int tid = threadIdx.x, lane = tid & 31, w = tid >> 5;
  int wm = w >> 2, wn = w & 3;
  int g = lane >> 4, ln = lane & 15;
  int e = blockIdx.z;
  int tokBase = blockIdx.y * BM;
  int n0 = blockIdx.x * BN;

  const __bf16* he = h  + (size_t)e * T_ * H_;
  const float*  wd = Wd + (size_t)e * H_ * D_;

  // per-thread A-copy unit: first 8 elements (16B) of one swizzle granule
  int arow = tid >> 2;
  int akg  = (tid & 3) * 16;
  unsigned aldsOff = (unsigned)(uintptr_t)(lsA + aidx(arow, akg));

  v8f acc[2][2] = {};

  for (int k0 = 0; k0 < H_; k0 += BK) {
    __syncthreads();
    // A: async DMA copy, 64 rows x 4 x 8 bf16, first half of each granule (one b128 per thread)
    {
      const __bf16* gp = he + (size_t)(tokBase + arow) * H_ + k0 + akg;
      asm volatile("global_load_async_to_lds_b128 %0, %1, off"
                   :: "v"(aldsOff), "v"(gp) : "memory");
    }
    // B: 64k x 128n fp32 -> bf16, packed, swizzled, conflict-free stores
#pragma unroll
    for (int j = 0; j < 4; ++j) {
      int idx   = tid + j * 256;
      int n4w   = idx & 3;
      int kpw   = (idx >> 2) & 7;
      int kpblk = (idx >> 5) & 3;
      int n4blk = idx >> 7;
      int k = 2 * (kpw + 8 * kpblk);
      int n = 4 * (n4w + 4 * n4blk);
      size_t off = (size_t)(k0 + k) * D_ + n0 + n;
      float4 v0 = *(const float4*)(wd + off);
      float4 v1 = *(const float4*)(wd + off + D_);
      if (k0 + BK < H_) __builtin_prefetch(wd + off + (size_t)BK * D_, 0, 0);
      *(unsigned*)(lsB + bidx(n + 0, k)) = pk2(v0.x, v1.x);
      *(unsigned*)(lsB + bidx(n + 1, k)) = pk2(v0.y, v1.y);
      *(unsigned*)(lsB + bidx(n + 2, k)) = pk2(v0.z, v1.z);
      *(unsigned*)(lsB + bidx(n + 3, k)) = pk2(v0.w, v1.w);
    }
    asm volatile("s_wait_asynccnt 0x0" ::: "memory");
    __syncthreads();
#pragma unroll
    for (int ks = 0; ks < BK; ks += 32) {
      v16bf a0 = ld_a(lsA, wm * 32 + ln,      ks, g);
      v16bf a1 = ld_a(lsA, wm * 32 + 16 + ln, ks, g);
#pragma unroll
      for (int ni = 0; ni < 2; ++ni) {
        v16bf b = ld_b(lsB, wn * 32 + ni * 16 + ln, ks, g);
        acc[0][ni] = wmma_bf16(a0, b, acc[0][ni]);
        acc[1][ni] = wmma_bf16(a1, b, acc[1][ni]);
      }
    }
  }

  // ---- LoRA rank stage ----
  const float* bd  = Bd + (size_t)e * R_ * D_;
  const float* hae = ha + ((size_t)e * T_ + tokBase) * R_;
  __syncthreads();
#pragma unroll
  for (int j = 0; j < 16; ++j) {
    int idx = tid + j * 256;
    int nn = idx >> 5, kk = idx & 31;
    lsB[bidx(nn, kk)] = f2bf(kk < R_ ? bd[(size_t)kk * D_ + n0 + nn] : 0.0f);
  }
#pragma unroll
  for (int j = 0; j < 8; ++j) {
    int idx = tid + j * 256;
    int row = idx >> 5, kk = idx & 31;
    lsA[aidx(row, kk)] = f2bf(kk < R_ ? hae[row * R_ + kk] : 0.0f);
  }
  __syncthreads();
  {
    v16bf a0 = ld_a(lsA, wm * 32 + ln,      0, g);
    v16bf a1 = ld_a(lsA, wm * 32 + 16 + ln, 0, g);
#pragma unroll
    for (int ni = 0; ni < 2; ++ni) {
      v16bf b = ld_b(lsB, wn * 32 + ni * 16 + ln, 0, g);
      acc[0][ni] = wmma_bf16(a0, b, acc[0][ni]);
      acc[1][ni] = wmma_bf16(a1, b, acc[1][ni]);
    }
  }

#pragma unroll
  for (int mi = 0; mi < 2; ++mi)
#pragma unroll
    for (int ni = 0; ni < 2; ++ni)
#pragma unroll
      for (int vg = 0; vg < 8; ++vg) {
        int m   = vg + 8 * g;
        int tok = tokBase + wm * 32 + mi * 16 + m;
        int col = n0 + wn * 32 + ni * 16 + ln;
        out[((size_t)e * T_ + tok) * D_ + col] = acc[mi][ni][vg];
      }
}

// ---------------------------------------------------------------------------
extern "C" void kernel_launch(void* const* d_in, const int* in_sizes, int n_in,
                              void* d_out, int out_size, void* d_ws, size_t ws_size,
                              hipStream_t stream) {
  (void)in_sizes; (void)n_in; (void)out_size; (void)ws_size;
  const float* x  = (const float*)d_in[0];
  // d_in[1] = num_tokens_per_expert (equal splits by construction; unused)
  const float* Wg = (const float*)d_in[2];
  const float* Wu = (const float*)d_in[3];
  const float* Wd = (const float*)d_in[4];
  const float* Ag = (const float*)d_in[5];
  const float* Bg = (const float*)d_in[6];
  const float* Au = (const float*)d_in[7];
  const float* Bu = (const float*)d_in[8];
  const float* Ad = (const float*)d_in[9];
  const float* Bd = (const float*)d_in[10];
  float* out = (float*)d_out;

  char* ws = (char*)d_ws;
  __bf16* h  = (__bf16*)ws;                                   // 64 MB
  float* xag = (float*)(ws + (size_t)E_ * T_ * H_ * 2);       // 512 KB
  float* xau = xag + (size_t)E_ * T_ * R_;                    // 512 KB
  float* ha  = xau + (size_t)E_ * T_ * R_;                    // 512 KB

  lora_xa_kernel<<<64, 256, 0, stream>>>(x, Ag, Au, xag, xau);
  gateup_kernel<<<dim3(H_ / BN, T_ / BM, E_), 256, 0, stream>>>(
      x, Wg, Wu, Bg, Bu, xag, xau, h);
  lora_ha_kernel<<<64, 256, 0, stream>>>(h, Ad, ha);
  down_kernel<<<dim3(D_ / BN, T_ / BM, E_), 256, 0, stream>>>(h, Wd, Bd, ha, out);
}
